// Network_66159676228094
// MI455X (gfx1250) — matrix operand order repaired and checked by reading, not compiled, not run on hardware
//
#include <hip/hip_runtime.h>
#include <hip/hip_bf16.h>
#include <cstdint>

#define DEV __device__ __forceinline__
#define USE_ASYNC_STAGE 1   // CDNA5 global_load_async_to_lds_b128 + s_wait_asynccnt path

// ---------------- problem constants ----------------
constexpr int BN = 64, SEQ = 720, PRED = 336, CIN = 321;
constexpr int PERIOD = 24, DMODEL = 128, SEGX = 30, SEGY = 14, KPOOL = 25, SSMK = 7;
constexpr int ROWS  = BN * CIN;        // 20544
constexpr int MSEAS = ROWS * PERIOD;   // 493056
constexpr int KTP = 736;               // 720 padded to mult of 32
constexpr int K8  = 352;               // 336 padded to mult of 32
constexpr int N5  = 2 * PRED;          // 672

// ---------------- workspace layout (bytes) ----------------
constexpr size_t OFF_TP  = 0;                                  // tp bf16 [R,736]; later reused as t2 bf16 [R,672]
constexpr size_t SZ_TP   = (size_t)ROWS * KTP * 2;
constexpr size_t OFF_R1  = OFF_TP + SZ_TP;                     // X bf16 [MSEAS,32] -> tt1 f32 [R,672] -> t4 f32 [R,336]
constexpr size_t SZ_R1   = (size_t)ROWS * N5 * 4;
constexpr size_t OFF_YS  = OFF_R1 + SZ_R1;                     // y_seas f32 [R,336]
constexpr size_t SZ_YS   = (size_t)ROWS * PRED * 4;
constexpr size_t OFF_T3  = OFF_YS + SZ_YS;                     // t3 bf16 [R,352]
constexpr size_t SZ_T3   = (size_t)ROWS * K8 * 2;
constexpr size_t OFF_W1B = OFF_T3 + SZ_T3;                     // w1b bf16 [32,128]
constexpr size_t OFF_W2B = OFF_W1B + 32 * 128 * 2;             // w2b bf16 [128,16]
constexpr size_t OFF_F5B = OFF_W2B + 128 * 16 * 2;             // fc5b bf16 [736,672]
constexpr size_t OFF_F7B = OFF_F5B + (size_t)KTP * N5 * 2;     // fc7b bf16 [672,336]
constexpr size_t OFF_F8B = OFF_F7B + (size_t)N5 * PRED * 2;    // fc8b bf16 [352,336]

// ---------------- types ----------------
typedef __bf16    v16bf __attribute__((ext_vector_type(16)));
typedef float     v8f   __attribute__((ext_vector_type(8)));
typedef uint32_t  v4u   __attribute__((ext_vector_type(4)));

union Frag { v16bf v; v4u q[2]; };

DEV unsigned short f2bf(float f) {
  union { float f; uint32_t u; } x; x.f = f;
  uint32_t r = x.u + 0x7FFFu + ((x.u >> 16) & 1u);   // round-to-nearest-even
  return (unsigned short)(r >> 16);
}
DEV float gelu_exact(float x) { return 0.5f * x * (1.0f + erff(x * 0.7071067811865476f)); }

// A fragment for v_wmma_f32_16x16x32_bf16: lane holds row m, K chunks [k0+hi*8,+8) and [k0+16+hi*8,+8)
DEV Frag load_a_frag(const unsigned short* __restrict__ A, int lda, int m, int k0, int hi) {
  const unsigned short* base = A + (size_t)m * lda + k0 + hi * 8;
  Frag f;
  f.q[0] = *reinterpret_cast<const v4u*>(base);
  f.q[1] = *reinterpret_cast<const v4u*>(base + 16);
  return f;
}
// B fragment from global: lane holds column n0+lo, K rows [k+hi*8,+8) and [k+16+hi*8,+8)
DEV Frag load_b_frag(const unsigned short* __restrict__ Bm, int ldb, int k, int n0) {
  int ln = threadIdx.x & 31, lo = ln & 15, hi = ln >> 4;
  const unsigned short* base = Bm + (size_t)(k + hi * 8) * ldb + n0 + lo;
  Frag f;
  unsigned short* e = reinterpret_cast<unsigned short*>(&f);
  #pragma unroll
  for (int s = 0; s < 8; ++s) { e[s] = base[(size_t)s * ldb]; e[s + 8] = base[(size_t)(s + 16) * ldb]; }
  return f;
}

// ---------------- weight pad/convert: f32 [sr,sc] -> bf16 [dr,dc], zero pad ----------------
__global__ void k_padcopy_bf16(unsigned short* __restrict__ dst, const float* __restrict__ src,
                               int sr, int sc, int dr, int dc) {
  int n = dr * dc;
  for (int idx = blockIdx.x * blockDim.x + threadIdx.x; idx < n; idx += gridDim.x * blockDim.x) {
    int rr = idx / dc, cc = idx - rr * dc;
    float v = (rr < sr && cc < sc) ? src[rr * sc + cc] : 0.0f;
    dst[idx] = f2bf(v);
  }
}

__global__ void k_zero_u32(uint32_t* __restrict__ p, int n) {
  for (int idx = blockIdx.x * blockDim.x + threadIdx.x; idx < n; idx += gridDim.x * blockDim.x)
    p[idx] = 0u;
}

// ---------------- seasonal prep: conv25 + avgpool25 + residual, segment-transposed bf16 X ----------------
__global__ void k_prep_seasonal(const float* __restrict__ s, const float* __restrict__ conv_w,
                                unsigned short* __restrict__ X) {
  int r = blockIdx.x, tid = threadIdx.x;
  int b = r / CIN, c = r - b * CIN;
  __shared__ float xs[SEQ];
  __shared__ float wc[KPOOL];
  for (int i = tid; i < SEQ; i += blockDim.x) xs[i] = s[(size_t)(b * SEQ + i) * CIN + c];
  if (tid < KPOOL) wc[tid] = conv_w[tid] + (1.0f / KPOOL);   // conv + avgpool share taps
  __syncthreads();
  for (int i = tid; i < SEQ; i += blockDim.x) {
    float acc = xs[i];
    #pragma unroll
    for (int k = 0; k < KPOOL; ++k) {
      int idx = i + k - KPOOL / 2;
      float xv = (idx >= 0 && idx < SEQ) ? xs[idx] : 0.0f;
      acc += xv * wc[k];
    }
    int seg = i / PERIOD, ph = i - seg * PERIOD;
    X[(size_t)(r * PERIOD + ph) * 32 + seg] = f2bf(acc);
  }
  for (int j = tid; j < PERIOD * 2; j += blockDim.x) {
    int ph = j >> 1, seg = SEGX + (j & 1);
    X[(size_t)(r * PERIOD + ph) * 32 + seg] = 0;
  }
}

// ---------------- trend prep: transpose t to [R,736] bf16, zero pad K ----------------
__global__ void k_prep_trend(const float* __restrict__ t, unsigned short* __restrict__ tp) {
  int r = blockIdx.x, tid = threadIdx.x;
  int b = r / CIN, c = r - b * CIN;
  for (int i = tid; i < KTP; i += blockDim.x) {
    float v = (i < SEQ) ? t[(size_t)(b * SEQ + i) * CIN + c] : 0.0f;
    tp[(size_t)r * KTP + i] = f2bf(v);
  }
}

// ---------------- fused seasonal MLP: [16,32]@[32,128] -> gelu -> (via LDS) @[128,16] ----------------
__global__ void k_seasonal_mlp(const unsigned short* __restrict__ X,
                               const unsigned short* __restrict__ w1b, const float* __restrict__ b1,
                               const unsigned short* __restrict__ w2b, const float* __restrict__ b2,
                               float* __restrict__ yseas) {
  int lane = threadIdx.x & 31, wave = threadIdx.x >> 5;
  int lo = lane & 15, hi = lane >> 4;
  int m0 = blockIdx.x * 64 + wave * 16;
  __shared__ __align__(16) unsigned short hl[4][16][DMODEL];

  Frag a = load_a_frag(X, 32, m0 + lo, 0, hi);
  #pragma unroll
  for (int nt = 0; nt < 8; ++nt) {
    Frag bfr = load_b_frag(w1b, DMODEL, 0, nt * 16);
    v8f acc = {};
    acc = __builtin_amdgcn_wmma_f32_16x16x32_bf16(false, a.v, false, bfr.v, (short)0, acc, false, false);
    float bv = b1[nt * 16 + lo];
    #pragma unroll
    for (int j = 0; j < 8; ++j) {
      float v = gelu_exact(acc[j] + bv);
      hl[wave][j + hi * 8][nt * 16 + lo] = f2bf(v);    // D-layout -> dense row-major in LDS
    }
  }
  __syncthreads();

  v8f accY = {};
  #pragma unroll
  for (int kt = 0; kt < 4; ++kt) {
    Frag a2;
    const unsigned short* hb = &hl[wave][lo][kt * 32 + hi * 8];
    a2.q[0] = *reinterpret_cast<const v4u*>(hb);
    a2.q[1] = *reinterpret_cast<const v4u*>(hb + 16);
    Frag b2f = load_b_frag(w2b, 16, kt * 32, 0);
    accY = __builtin_amdgcn_wmma_f32_16x16x32_bf16(false, a2.v, false, b2f.v, (short)0, accY, false, false);
  }
  if (lo < SEGY) {
    float bv = b2[lo];
    #pragma unroll
    for (int j = 0; j < 8; ++j) {
      int m = m0 + j + hi * 8;
      int r = m / PERIOD, ph = m - r * PERIOD;
      yseas[(size_t)r * PRED + lo * PERIOD + ph] = accY[j] + bv;  // y[b, seg*24+ph, c] row
    }
  }
}

// ---------------- GEMM: 4 waves (M tile 64), wave = 16 x (NSUB*16) ----------------
// B tile double-buffered in LDS, filled with CDNA5 async global->LDS DMA (ASYNCcnt).
// EPI: 0 = f32 store (+bias), 1 = gelu f32 store (+bias), 2 = bf16 store (+bias)
template <int NSUB, int EPI>
__global__ void k_gemm_bf16(const unsigned short* __restrict__ A, int lda,
                            const unsigned short* __restrict__ Bw, int ldb,
                            const float* __restrict__ bias,
                            void* __restrict__ Cout, int ldc, int K) {
  constexpr int NB  = NSUB * 16;
  constexpr int BSP = NB + 8;                 // padded LDS row; (NB+8)*2 bytes is 16B multiple
  __shared__ __align__(16) unsigned short Bs[2][32][BSP];
  const int tid  = threadIdx.x;
  const int lane = tid & 31, wave = tid >> 5;
  const int lo = lane & 15, hi = lane >> 4;
  const int m0 = blockIdx.x * 64 + wave * 16;
  const int n0 = blockIdx.y * NB;
  const int mrow = m0 + lo;

  v8f acc[NSUB];
  #pragma unroll
  for (int u = 0; u < NSUB; ++u) { v8f z = {}; acc[u] = z; }

  // Stage B tile [32 x NB] for K rows [kbase, kbase+32) into Bs[bsel] (32B chunks).
  auto stage = [&](int kbase, int bsel) {
    for (int idx = tid; idx < 32 * NSUB; idx += 128) {
      int row = idx / NSUB, ch = idx - row * NSUB;
      const unsigned short* gp = Bw + (size_t)(kbase + row) * ldb + n0 + ch * 16;
      unsigned short* lp = &Bs[bsel][row][ch * 16];
#if USE_ASYNC_STAGE
      uint64_t ga = (uint64_t)(uintptr_t)gp;
      uint32_t la = (uint32_t)(uintptr_t)lp;      // generic LDS addr: low 32 bits == LDS offset
      asm volatile("global_load_async_to_lds_b128 %0, %1, off\n\t"
                   "global_load_async_to_lds_b128 %2, %3, off"
                   :: "v"(la), "v"(ga), "v"(la + 16u), "v"(ga + 16ull)
                   : "memory");
#else
      v4u q0 = *reinterpret_cast<const v4u*>(gp);
      v4u q1 = *reinterpret_cast<const v4u*>(gp + 8);
      *reinterpret_cast<v4u*>(lp) = q0;
      *reinterpret_cast<v4u*>(lp + 8) = q1;
#endif
    }
  };

  stage(0, 0);                                  // prologue fill of buffer 0
  int buf = 0;
  for (int k0 = 0; k0 < K; k0 += 32) {
#if USE_ASYNC_STAGE
    asm volatile("s_wait_asynccnt 0x0" ::: "memory");   // my async fills of current buffer done
#endif
    __syncthreads();                            // everyone's fills visible; prev reads of other buf done
    if (k0 + 32 < K) {
      stage(k0 + 32, buf ^ 1);                  // overlap next-tile DMA with WMMA below
      __builtin_prefetch(A + (size_t)mrow * lda + k0 + 32, 0, 1);
    }
    Frag a = load_a_frag(A, lda, mrow, k0, hi);
    #pragma unroll
    for (int u = 0; u < NSUB; ++u) {
      Frag b;
      unsigned short* be = reinterpret_cast<unsigned short*>(&b);
      #pragma unroll
      for (int s = 0; s < 8; ++s) { be[s] = Bs[buf][hi * 8 + s][u * 16 + lo]; be[s + 8] = Bs[buf][16 + hi * 8 + s][u * 16 + lo]; }
      acc[u] = __builtin_amdgcn_wmma_f32_16x16x32_bf16(false, a.v, false, b.v, (short)0, acc[u], false, false);
    }
    buf ^= 1;
  }

  #pragma unroll
  for (int u = 0; u < NSUB; ++u) {
    int n = n0 + u * 16 + lo;
    float bv = bias[n];
    #pragma unroll
    for (int j = 0; j < 8; ++j) {
      int m = m0 + j + hi * 8;
      float v = acc[u][j] + bv;
      if (EPI == 1) v = gelu_exact(v);
      if (EPI == 2) ((unsigned short*)Cout)[(size_t)m * ldc + n] = f2bf(v);
      else          ((float*)Cout)[(size_t)m * ldc + n] = v;
    }
  }
}

// ---------------- layernorm over 672, write bf16 ----------------
__global__ void k_layernorm(const float* __restrict__ tt1, const float* __restrict__ g,
                            const float* __restrict__ bta, unsigned short* __restrict__ t2) {
  int r = blockIdx.x, tid = threadIdx.x;
  __shared__ float xrow[N5];
  __shared__ float red[256];
  float s1 = 0.f;
  for (int i = tid; i < N5; i += blockDim.x) { float v = tt1[(size_t)r * N5 + i]; xrow[i] = v; s1 += v; }
  red[tid] = s1; __syncthreads();
  for (int off = 128; off > 0; off >>= 1) { if (tid < off) red[tid] += red[tid + off]; __syncthreads(); }
  float mu = red[0] * (1.0f / N5);
  __syncthreads();
  float s2 = 0.f;
  for (int i = tid; i < N5; i += blockDim.x) { float d = xrow[i] - mu; s2 += d * d; }
  red[tid] = s2; __syncthreads();
  for (int off = 128; off > 0; off >>= 1) { if (tid < off) red[tid] += red[tid + off]; __syncthreads(); }
  float rstd = rsqrtf(red[0] * (1.0f / N5) + 1e-5f);
  for (int i = tid; i < N5; i += blockDim.x)
    t2[(size_t)r * N5 + i] = f2bf((xrow[i] - mu) * rstd * g[i] + bta[i]);
}

// ---------------- combine: SSM depthwise conv on seasonal + trend add, final transpose ----------------
__global__ void k_combine(const float* __restrict__ t4, const float* __restrict__ yseas,
                          const float* __restrict__ ssmA, const float* __restrict__ ssmB,
                          const float* __restrict__ ssmC, float* __restrict__ out) {
  int r = blockIdx.x, tid = threadIdx.x;
  int b = r / CIN, c = r - b * CIN;
  __shared__ float ys[PRED];
  for (int p = tid; p < PRED; p += blockDim.x) ys[p] = yseas[(size_t)r * PRED + p];
  float Av = ssmA[c], Bv = ssmB[c], Cv = ssmC[c];
  float kern[SSMK];
  float pk = 1.f;
  #pragma unroll
  for (int k = 0; k < SSMK; ++k) { kern[k] = Cv * pk * Bv; pk *= Av; }
  __syncthreads();
  for (int p = tid; p < PRED; p += blockDim.x) {
    float acc = t4[(size_t)r * PRED + p];
    #pragma unroll
    for (int k = 0; k < SSMK; ++k) {
      int q = p + k - SSMK / 2;
      if (q >= 0 && q < PRED) acc += kern[k] * ys[q];
    }
    out[((size_t)b * PRED + p) * CIN + c] = acc;
  }
}

// ---------------- host launcher ----------------
extern "C" void kernel_launch(void* const* d_in, const int* in_sizes, int n_in,
                              void* d_out, int out_size, void* d_ws, size_t ws_size,
                              hipStream_t stream) {
  const float* s      = (const float*)d_in[0];
  const float* t      = (const float*)d_in[1];
  const float* conv_w = (const float*)d_in[2];
  const float* mlp_w1 = (const float*)d_in[3];
  const float* mlp_b1 = (const float*)d_in[4];
  const float* mlp_w2 = (const float*)d_in[5];
  const float* mlp_b2 = (const float*)d_in[6];
  const float* ssmA   = (const float*)d_in[7];
  const float* ssmB   = (const float*)d_in[8];
  const float* ssmC   = (const float*)d_in[9];
  const float* fc5_w  = (const float*)d_in[10];
  const float* fc5_b  = (const float*)d_in[11];
  const float* ln_g   = (const float*)d_in[12];
  const float* ln_b   = (const float*)d_in[13];
  const float* fc7_w  = (const float*)d_in[14];
  const float* fc7_b  = (const float*)d_in[15];
  const float* fc8_w  = (const float*)d_in[16];
  const float* fc8_b  = (const float*)d_in[17];
  float* out = (float*)d_out;

  char* ws = (char*)d_ws;
  unsigned short* tp  = (unsigned short*)(ws + OFF_TP);
  unsigned short* t2  = (unsigned short*)(ws + OFF_TP);   // reuse after tp consumed
  unsigned short* X   = (unsigned short*)(ws + OFF_R1);
  float*          tt1 = (float*)(ws + OFF_R1);            // reuse after X consumed
  float*          t4  = (float*)(ws + OFF_R1);            // reuse after tt1 consumed
  float*          yse = (float*)(ws + OFF_YS);
  unsigned short* t3  = (unsigned short*)(ws + OFF_T3);
  unsigned short* w1b = (unsigned short*)(ws + OFF_W1B);
  unsigned short* w2b = (unsigned short*)(ws + OFF_W2B);
  unsigned short* f5b = (unsigned short*)(ws + OFF_F5B);
  unsigned short* f7b = (unsigned short*)(ws + OFF_F7B);
  unsigned short* f8b = (unsigned short*)(ws + OFF_F8B);

  // --- weight conversion / padding (zero pads make WMMA K-loops maskless) ---
  k_padcopy_bf16<<<16,   256, 0, stream>>>(w1b, mlp_w1, SEGX, DMODEL, 32, DMODEL);
  k_padcopy_bf16<<<8,    256, 0, stream>>>(w2b, mlp_w2, DMODEL, SEGY, DMODEL, 16);
  k_padcopy_bf16<<<1932, 256, 0, stream>>>(f5b, fc5_w, SEQ, N5, KTP, N5);
  k_padcopy_bf16<<<882,  256, 0, stream>>>(f7b, fc7_w, N5, PRED, N5, PRED);
  k_padcopy_bf16<<<462,  256, 0, stream>>>(f8b, fc8_w, PRED, PRED, K8, PRED);
  {
    int nT3 = (int)(SZ_T3 / 4);
    k_zero_u32<<<(nT3 + 255) / 256, 256, 0, stream>>>((uint32_t*)t3, nT3);  // zero t3 incl. K-pad cols
  }

  // --- seasonal branch ---
  k_prep_seasonal<<<ROWS, 256, 0, stream>>>(s, conv_w, X);
  k_seasonal_mlp<<<MSEAS / 64, 128, 0, stream>>>(X, w1b, mlp_b1, w2b, mlp_b2, yse);

  // --- trend branch ---
  k_prep_trend<<<ROWS, 256, 0, stream>>>(t, tp);
  k_gemm_bf16<6, 1><<<dim3(ROWS / 64, N5 / 96), 128, 0, stream>>>(tp, KTP, f5b, N5, fc5_b, tt1, N5, KTP);
  k_layernorm<<<ROWS, 256, 0, stream>>>(tt1, ln_g, ln_b, t2);
  k_gemm_bf16<3, 2><<<dim3(ROWS / 64, PRED / 48), 128, 0, stream>>>(t2, N5, f7b, PRED, fc7_b, t3, K8, N5);
  k_gemm_bf16<3, 0><<<dim3(ROWS / 64, PRED / 48), 128, 0, stream>>>(t3, K8, f8b, PRED, fc8_b, t4, PRED, K8);

  // --- SSM conv + add + transpose to [B, P, C] ---
  k_combine<<<ROWS, 256, 0, stream>>>(t4, yse, ssmA, ssmB, ssmC, out);
}
